// GATEncoder_21586505630141
// MI455X (gfx1250) — compile-verified, matrix operand order checked by reading, not run on hardware
//
#include <hip/hip_runtime.h>
#include <hip/hip_bf16.h>

typedef __bf16 bf16;
typedef __attribute__((ext_vector_type(16))) __bf16 v16bf;
typedef __attribute__((ext_vector_type(8)))  float  v8f;

// ---------------------------------------------------------------- utilities

__device__ __forceinline__ void atomicMaxF(float* addr, float val) {
  if (val >= 0.0f) atomicMax((int*)addr, __float_as_int(val));
  else             atomicMin((unsigned int*)addr, (unsigned int)__float_as_int(val));
}

__global__ void k_fillf(float* p, float v, int n) {
  int i = blockIdx.x * blockDim.x + threadIdx.x;
  if (i < n) p[i] = v;
}

__global__ void k_copyi(const int* __restrict__ a, int* __restrict__ b, int n) {
  int i = blockIdx.x * blockDim.x + threadIdx.x;
  if (i < n) b[i] = a[i];
}

// ------------------------------------------------ pack W1 into WMMA B-fragment layout
// layout: wp[(colTile*kChunks + chunk)*32 + lane] = 16 contiguous bf16 (32B) for that lane.
// lane<16: col=colTile*16+lane, K = k0..k0+7, k0+16..k0+23 (k0=chunk*32)
// lane>=16: col=colTile*16+lane-16, K = k0+8..k0+15, k0+24..k0+31
__global__ void k_packw(const float* __restrict__ W, v16bf* __restrict__ wp,
                        int F, int Ncols, int kChunks) {
  int t = blockIdx.x * blockDim.x + threadIdx.x;
  int total = (Ncols / 16) * kChunks * 32;
  if (t >= total) return;
  int lane = t & 31;
  int ch = (t >> 5) % kChunks;
  int colTile = (t >> 5) / kChunks;
  int col = colTile * 16 + (lane & 15);
  int k0 = ch * 32 + (lane >> 4) * 8;
  v16bf v;
#pragma unroll
  for (int j = 0; j < 8; ++j) {
    int ka = k0 + j, kb = k0 + 16 + j;
    float va = W[(size_t)(ka < F ? ka : 0) * Ncols + col];
    float vb = W[(size_t)(kb < F ? kb : 0) * Ncols + col];
    v[j]     = (bf16)(ka < F ? va : 0.0f);
    v[8 + j] = (bf16)(kb < F ? vb : 0.0f);
  }
  wp[t] = v;
}

// ------------------------------------------------ A-fragment load (full chunk, no guards)
__device__ __forceinline__ v16bf loadA(const float* __restrict__ ap, int k0) {
  const float4* p0 = (const float4*)(ap + k0);
  float4 q0 = p0[0], q1 = p0[1];
  const float4* p1 = (const float4*)(ap + k0 + 16);
  float4 q2 = p1[0], q3 = p1[1];
  v16bf a;
  a[0]=(bf16)q0.x;  a[1]=(bf16)q0.y;  a[2]=(bf16)q0.z;  a[3]=(bf16)q0.w;
  a[4]=(bf16)q1.x;  a[5]=(bf16)q1.y;  a[6]=(bf16)q1.z;  a[7]=(bf16)q1.w;
  a[8]=(bf16)q2.x;  a[9]=(bf16)q2.y;  a[10]=(bf16)q2.z; a[11]=(bf16)q2.w;
  a[12]=(bf16)q3.x; a[13]=(bf16)q3.y; a[14]=(bf16)q3.z; a[15]=(bf16)q3.w;
  return a;
}

// A-fragment load for the tail chunk: clamped loads + value select (v_cndmask, no exec juggling)
__device__ __forceinline__ v16bf loadA_tail(const float* __restrict__ ap, int k0, int F) {
  v16bf a;
#pragma unroll
  for (int j = 0; j < 8; ++j) {
    int ka = k0 + j, kb = k0 + 16 + j;
    float va = ap[ka < F ? ka : 0];
    float vb = ap[kb < F ? kb : 0];
    a[j]     = (bf16)(ka < F ? va : 0.0f);
    a[8 + j] = (bf16)(kb < F ? vb : 0.0f);
  }
  return a;
}

#define WMMA_BF16(A, B, C) __builtin_amdgcn_wmma_f32_16x16x32_bf16(false, A, false, B, (short)0, C, false, false)

// ------------------------------------------------ GEMM1: h1[N,Ncols] = x[N,F] @ W1
// one wave computes a 32x64 strip of C: 2 A fragments share 4 B fragments -> 8 WMMAs/iter.
__global__ __launch_bounds__(256) void k_gemm1(const float* __restrict__ x,
                                               const v16bf* __restrict__ wp,
                                               float* __restrict__ h1,
                                               int N, int F, int kChunks,
                                               int colGroups, int Ncols) {
  int wid  = (blockIdx.x * blockDim.x + threadIdx.x) >> 5;
  int lane = threadIdx.x & 31;
  int rowBlocks = (N + 31) >> 5;
  int rowBlk = wid / colGroups;
  int cg     = wid % colGroups;
  if (rowBlk >= rowBlocks) return;

  int half = lane >> 4;
  int r0 = rowBlk * 32 + (lane & 15);
  int r1 = r0 + 16;
  const float* ap0 = x + (size_t)(r0 < N ? r0 : 0) * F;   // clamped: safe; garbage rows never stored
  const float* ap1 = x + (size_t)(r1 < N ? r1 : 0) * F;

  v8f c00 = {}, c01 = {}, c02 = {}, c03 = {};
  v8f c10 = {}, c11 = {}, c12 = {}, c13 = {};
  int bstride = kChunks * 32;
  const v16bf* wpb = wp + (size_t)(cg * 4) * bstride + lane;

  int fullCh = F >> 5;   // chunks whose 32 k-values are all in range (uniform!)
  int ch = 0;
  for (; ch < fullCh; ++ch) {
    int k0 = ch * 32 + half * 8;
    v16bf a0 = loadA(ap0, k0);
    v16bf a1 = loadA(ap1, k0);
    const v16bf* wq = wpb + ch * 32;
    v16bf b0 = wq[0];
    v16bf b1 = wq[bstride];
    v16bf b2 = wq[2 * bstride];
    v16bf b3 = wq[3 * bstride];
    c00 = WMMA_BF16(a0, b0, c00);
    c10 = WMMA_BF16(a1, b0, c10);
    c01 = WMMA_BF16(a0, b1, c01);
    c11 = WMMA_BF16(a1, b1, c11);
    c02 = WMMA_BF16(a0, b2, c02);
    c12 = WMMA_BF16(a1, b2, c12);
    c03 = WMMA_BF16(a0, b3, c03);
    c13 = WMMA_BF16(a1, b3, c13);
  }
  if (ch < kChunks) {    // single tail chunk, zero-padded in both A (here) and packed W
    int k0 = ch * 32 + half * 8;
    v16bf a0 = loadA_tail(ap0, k0, F);
    v16bf a1 = loadA_tail(ap1, k0, F);
    const v16bf* wq = wpb + ch * 32;
    v16bf b0 = wq[0];
    v16bf b1 = wq[bstride];
    v16bf b2 = wq[2 * bstride];
    v16bf b3 = wq[3 * bstride];
    c00 = WMMA_BF16(a0, b0, c00);
    c10 = WMMA_BF16(a1, b0, c10);
    c01 = WMMA_BF16(a0, b1, c01);
    c11 = WMMA_BF16(a1, b1, c11);
    c02 = WMMA_BF16(a0, b2, c02);
    c12 = WMMA_BF16(a1, b2, c12);
    c03 = WMMA_BF16(a0, b3, c03);
    c13 = WMMA_BF16(a1, b3, c13);
  }

  int col0 = cg * 64 + (lane & 15);
#pragma unroll
  for (int i = 0; i < 8; ++i) {
    int ra = rowBlk * 32 + half * 8 + i;
    if (ra < N) {
      float* hp = h1 + (size_t)ra * Ncols + col0;
      hp[0] = c00[i]; hp[16] = c01[i]; hp[32] = c02[i]; hp[48] = c03[i];
    }
    int rb = ra + 16;
    if (rb < N) {
      float* hp = h1 + (size_t)rb * Ncols + col0;
      hp[0] = c10[i]; hp[16] = c11[i]; hp[32] = c12[i]; hp[48] = c13[i];
    }
  }
}

// ------------------------------------------------ attention coefficients (layer 1)
__global__ __launch_bounds__(256) void k_att1(const float* __restrict__ h1,
                                              const float* __restrict__ as,
                                              const float* __restrict__ ad,
                                              float* __restrict__ oas,
                                              float* __restrict__ oad,
                                              int N, int H, int C) {
  int wid  = (blockIdx.x * blockDim.x + threadIdx.x) >> 5;
  int lane = threadIdx.x & 31;
  int n = wid / H, h = wid % H;
  if (n >= N) return;
  const float* hp = h1 + (size_t)n * H * C + (size_t)h * C;
  float s = 0.f, d = 0.f;
  for (int c = lane; c < C; c += 32) {
    float v = hp[c];
    s += v * as[h * C + c];
    d += v * ad[h * C + c];
  }
  for (int m = 16; m >= 1; m >>= 1) { s += __shfl_xor(s, m, 32); d += __shfl_xor(d, m, 32); }
  if (lane == 0) { oas[n * H + h] = s; oad[n * H + h] = d; }
}

// ------------------------------------------------ CSR build (edges incl. self loops)
__global__ void k_count(const int* __restrict__ ei, int E, int N, int* __restrict__ deg) {
  int e = blockIdx.x * blockDim.x + threadIdx.x;
  if (e >= E + N) return;
  int d = (e < E) ? ei[E + e] : (e - E);
  atomicAdd(&deg[d], 1);
}

__global__ void k_scan(const int* __restrict__ deg, int* __restrict__ offv, int n) {
  __shared__ int buf[1024];
  __shared__ int carry;
  if (threadIdx.x == 0) { carry = 0; offv[0] = 0; }
  __syncthreads();
  for (int base = 0; base < n; base += 1024) {
    int i = base + (int)threadIdx.x;
    int v = (i < n) ? deg[i] : 0;
    buf[threadIdx.x] = v;
    __syncthreads();
    for (int d = 1; d < 1024; d <<= 1) {
      int t = (threadIdx.x >= (unsigned)d) ? buf[threadIdx.x - d] : 0;
      __syncthreads();
      buf[threadIdx.x] += t;
      __syncthreads();
    }
    if (i < n) offv[i + 1] = carry + buf[threadIdx.x];
    __syncthreads();
    if (threadIdx.x == 0) carry += buf[1023];
    __syncthreads();
  }
}

__global__ void k_place(const int* __restrict__ ei, int E, int N,
                        int* __restrict__ cur, int* __restrict__ csr) {
  int e = blockIdx.x * blockDim.x + threadIdx.x;
  if (e >= E + N) return;
  int d = (e < E) ? ei[E + e] : (e - E);
  int p = atomicAdd(&cur[d], 1);
  csr[p] = e;
}

// ------------------------------------------------ edge softmax (generic over H)
__global__ void k_edgeA(const int* __restrict__ ei, int E, int N, int H,
                        const float* __restrict__ oas, const float* __restrict__ oad,
                        float* __restrict__ ev, float* __restrict__ m) {
  int e = blockIdx.x * blockDim.x + threadIdx.x;
  if (e >= E + N) return;
  int s = (e < E) ? ei[e] : (e - E);
  int d = (e < E) ? ei[E + e] : (e - E);
  for (int h = 0; h < H; ++h) {
    float v = oas[s * H + h] + oad[d * H + h];
    v = v > 0.f ? v : 0.2f * v;                 // LeakyReLU(0.2)
    ev[(size_t)e * H + h] = v;
    atomicMaxF(&m[d * H + h], v);
  }
}

__global__ void k_edgeB(const int* __restrict__ ei, int E, int N, int H,
                        const float* __restrict__ m, float* __restrict__ ev,
                        float* __restrict__ den) {
  int e = blockIdx.x * blockDim.x + threadIdx.x;
  if (e >= E + N) return;
  int d = (e < E) ? ei[E + e] : (e - E);
  for (int h = 0; h < H; ++h) {
    float ex = expf(ev[(size_t)e * H + h] - m[d * H + h]);
    ev[(size_t)e * H + h] = ex;
    atomicAdd(&den[d * H + h], ex);
  }
}

// ------------------------------------------------ layer-1 aggregation (H=4, C=512) + head-mean + bias
__global__ __launch_bounds__(256) void k_agg1(const float* __restrict__ h1,
                                              const float* __restrict__ ev,
                                              const float* __restrict__ den,
                                              const int* __restrict__ offv,
                                              const int* __restrict__ csr,
                                              const int* __restrict__ ei,
                                              const float* __restrict__ b1,
                                              float* __restrict__ hm, int N, int E) {
  int n    = (blockIdx.x * blockDim.x + threadIdx.x) >> 5;
  int lane = threadIdx.x & 31;
  if (n >= N) return;
  float acc[16];
#pragma unroll
  for (int j = 0; j < 16; ++j) acc[j] = 0.f;
  float inv[4];
#pragma unroll
  for (int h = 0; h < 4; ++h) inv[h] = 1.f / (den[n * 4 + h] + 1e-16f);
  int st = offv[n], en = offv[n + 1];
  for (int i = st; i < en; ++i) {
    int e = csr[i];
    int s = (e < E) ? ei[e] : (e - E);
    float4 a4 = *(const float4*)(ev + (size_t)e * 4);
    float av0 = a4.x * inv[0], av1 = a4.y * inv[1], av2 = a4.z * inv[2], av3 = a4.w * inv[3];
    const float* hp = h1 + (size_t)s * 2048 + lane * 16;
#pragma unroll
    for (int h = 0; h < 4; ++h) {
      float al = (h == 0) ? av0 : (h == 1) ? av1 : (h == 2) ? av2 : av3;
      const float4* p = (const float4*)(hp + h * 512);
      float4 v0 = p[0], v1 = p[1], v2 = p[2], v3 = p[3];
      acc[0]  += al * v0.x; acc[1]  += al * v0.y; acc[2]  += al * v0.z; acc[3]  += al * v0.w;
      acc[4]  += al * v1.x; acc[5]  += al * v1.y; acc[6]  += al * v1.z; acc[7]  += al * v1.w;
      acc[8]  += al * v2.x; acc[9]  += al * v2.y; acc[10] += al * v2.z; acc[11] += al * v2.w;
      acc[12] += al * v3.x; acc[13] += al * v3.y; acc[14] += al * v3.z; acc[15] += al * v3.w;
    }
  }
  float* op = hm + (size_t)n * 512 + lane * 16;
#pragma unroll
  for (int j = 0; j < 16; ++j) op[j] = acc[j] * 0.25f + b1[lane * 16 + j];
}

// ------------------------------------------------ BatchNorm stats + apply + ELU
__global__ __launch_bounds__(512) void k_bnstats(const float* __restrict__ hm,
                                                 float* __restrict__ bsum,
                                                 float* __restrict__ bsq, int N, int C) {
  int c = threadIdx.x;  // blockDim.x == C == 512
  float s = 0.f, s2 = 0.f;
  for (int r = blockIdx.x; r < N; r += gridDim.x) {
    float v = hm[(size_t)r * C + c];
    s += v; s2 += v * v;
  }
  atomicAdd(&bsum[c], s);
  atomicAdd(&bsq[c], s2);
}

__global__ void k_bnapply(float* __restrict__ hm, const float* __restrict__ bsum,
                          const float* __restrict__ bsq, const float* __restrict__ gamma,
                          const float* __restrict__ beta, int N, int C) {
  size_t t = (size_t)blockIdx.x * blockDim.x + threadIdx.x;
  if (t >= (size_t)N * C) return;
  int c = (int)(t % C);
  float invN = 1.f / (float)N;
  float mu  = bsum[c] * invN;
  float var = bsq[c] * invN - mu * mu;
  float y = (hm[t] - mu) * rsqrtf(var + 1e-5f) * gamma[c] + beta[c];
  hm[t] = y > 0.f ? y : expm1f(y);   // ELU
}

// ------------------------------------------------ GEMM2 (C=512 -> L=30) + layer-2 attention dots
__global__ __launch_bounds__(256) void k_gemm2(const float* __restrict__ hm,
                                               const float* __restrict__ W2,
                                               const float* __restrict__ as2,
                                               const float* __restrict__ ad2,
                                               float* __restrict__ h2,
                                               float* __restrict__ oas,
                                               float* __restrict__ oad,
                                               int N, int C, int L) {
  __shared__ float sh[8][512];
  int warp = threadIdx.x >> 5, lane = threadIdx.x & 31;
  int n = blockIdx.x * 8 + warp;
  bool act = (n < N);
  if (act) {
    const float4* p = (const float4*)(hm + (size_t)n * C + lane * 16);
    float4* q = (float4*)&sh[warp][lane * 16];
    q[0] = p[0]; q[1] = p[1]; q[2] = p[2]; q[3] = p[3];
  }
  __syncthreads();
  float ps = 0.f, pd = 0.f;
  if (act && lane < L) {
    float acc = 0.f;
#pragma unroll 4
    for (int k = 0; k < C; ++k) acc += sh[warp][k] * W2[k * L + lane];
    h2[(size_t)n * L + lane] = acc;
    ps = acc * as2[lane];
    pd = acc * ad2[lane];
  }
  for (int m = 16; m >= 1; m >>= 1) { ps += __shfl_xor(ps, m, 32); pd += __shfl_xor(pd, m, 32); }
  if (act && lane == 0) { oas[n] = ps; oad[n] = pd; }
}

// ------------------------------------------------ layer-2 aggregation (H=1, L=30) + bias
__global__ __launch_bounds__(256) void k_agg2(const float* __restrict__ h2,
                                              const float* __restrict__ ev,
                                              const float* __restrict__ den,
                                              const int* __restrict__ offv,
                                              const int* __restrict__ csr,
                                              const int* __restrict__ ei,
                                              const float* __restrict__ b2,
                                              float* __restrict__ out, int N, int E, int L) {
  int n    = (blockIdx.x * blockDim.x + threadIdx.x) >> 5;
  int lane = threadIdx.x & 31;
  if (n >= N) return;
  float acc = 0.f;
  float invd = 1.f / (den[n] + 1e-16f);
  int st = offv[n], en = offv[n + 1];
  for (int i = st; i < en; ++i) {
    int e = csr[i];
    int s = (e < E) ? ei[e] : (e - E);
    float a = ev[e] * invd;
    if (lane < L) acc += a * h2[(size_t)s * L + lane];
  }
  if (lane < L) out[(size_t)n * L + lane] = acc + b2[lane];
}

// ================================================================ host launcher
extern "C" void kernel_launch(void* const* d_in, const int* in_sizes, int n_in,
                              void* d_out, int out_size, void* d_ws, size_t ws_size,
                              hipStream_t stream) {
  const float* x    = (const float*)d_in[0];
  const int*   ei   = (const int*)d_in[1];
  const float* W1   = (const float*)d_in[2];
  const float* as1  = (const float*)d_in[3];
  const float* ad1  = (const float*)d_in[4];
  const float* b1   = (const float*)d_in[5];
  const float* gmma = (const float*)d_in[6];
  const float* beta = (const float*)d_in[7];
  const float* W2   = (const float*)d_in[8];
  const float* as2  = (const float*)d_in[9];
  const float* ad2  = (const float*)d_in[10];
  const float* b2   = (const float*)d_in[11];
  float* z = (float*)d_out;

  const int HID  = in_sizes[5];                 // 512
  const int H    = in_sizes[3] / HID;           // 4
  const int Nc   = H * HID;                     // 2048
  const int F    = in_sizes[2] / Nc;            // 2000
  const int N    = (int)((long long)in_sizes[0] / F);  // 50000
  const int E    = in_sizes[1] / 2;             // 800000
  const int L    = in_sizes[11];                // 30
  const int E2   = E + N;
  const int kCh  = (F + 31) / 32;               // 63
  const int colGroups = Nc / 64;                // 32
  const int colTiles  = Nc / 16;                // 128

  // workspace carve-out
  size_t off = 0;
  auto carve = [&](size_t bytes) -> char* {
    char* p = (char*)d_ws + off;
    off = (off + bytes + 255) & ~(size_t)255;
    return p;
  };
  v16bf* wp   = (v16bf*)carve((size_t)colTiles * kCh * 32 * sizeof(v16bf));
  float* h1   = (float*)carve((size_t)N * Nc * 4);
  float* oas1 = (float*)carve((size_t)N * H * 4);
  float* oad1 = (float*)carve((size_t)N * H * 4);
  float* m1   = (float*)carve((size_t)N * H * 4);
  float* den1 = (float*)carve((size_t)N * H * 4);
  float* e1   = (float*)carve((size_t)E2 * H * 4);
  int*   deg  = (int*)carve((size_t)N * 4);
  int*   offv = (int*)carve((size_t)(N + 1) * 4);
  int*   cur  = (int*)carve((size_t)N * 4);
  int*   csr  = (int*)carve((size_t)E2 * 4);
  float* hm   = (float*)carve((size_t)N * HID * 4);
  float* bsum = (float*)carve(512 * 4);
  float* bsq  = (float*)carve(512 * 4);
  float* h2   = (float*)carve((size_t)N * L * 4);
  float* oas2 = (float*)carve((size_t)N * 4);
  float* oad2 = (float*)carve((size_t)N * 4);
  float* m2   = (float*)carve((size_t)N * 4);
  float* den2 = (float*)carve((size_t)N * 4);
  float* e2   = (float*)carve((size_t)E2 * 4);
  (void)ws_size; (void)n_in; (void)out_size;

  const int T = 256;
  auto cdiv = [](long long a, long long b) { return (int)((a + b - 1) / b); };

  // 1) pack W1 (bf16, B-fragment layout; L2-resident afterwards)
  k_packw<<<cdiv((long long)colTiles * kCh * 32, T), T, 0, stream>>>(W1, wp, F, Nc, kCh);

  // 2) main GEMM via WMMA bf16 -> f32 (32x64 strip per wave)
  {
    long long waves = (long long)((N + 31) / 32) * colGroups;
    k_gemm1<<<cdiv(waves, 8), T, 0, stream>>>(x, wp, h1, N, F, kCh, colGroups, Nc);
  }

  // 3) attention dot products for layer 1
  k_att1<<<cdiv((long long)N * H, 8), T, 0, stream>>>(h1, as1, ad1, oas1, oad1, N, H, HID);

  // 4) CSR build over edges + self loops
  hipMemsetAsync(deg, 0, (size_t)N * 4, stream);
  k_count<<<cdiv(E2, T), T, 0, stream>>>(ei, E, N, deg);
  k_scan<<<1, 1024, 0, stream>>>(deg, offv, N);
  k_copyi<<<cdiv(N, T), T, 0, stream>>>(offv, cur, N);
  k_place<<<cdiv(E2, T), T, 0, stream>>>(ei, E, N, cur, csr);

  // 5) edge softmax, layer 1
  k_fillf<<<cdiv((long long)N * H, T), T, 0, stream>>>(m1, -INFINITY, N * H);
  hipMemsetAsync(den1, 0, (size_t)N * H * 4, stream);
  k_edgeA<<<cdiv(E2, T), T, 0, stream>>>(ei, E, N, H, oas1, oad1, e1, m1);
  k_edgeB<<<cdiv(E2, T), T, 0, stream>>>(ei, E, N, H, m1, e1, den1);

  // 6) aggregate + head-mean + bias
  k_agg1<<<cdiv((long long)N, 8), T, 0, stream>>>(h1, e1, den1, offv, csr, ei, b1, hm, N, E);

  // 7) BatchNorm (batch stats) + ELU, in place on hm
  hipMemsetAsync(bsum, 0, 512 * 4, stream);
  hipMemsetAsync(bsq, 0, 512 * 4, stream);
  k_bnstats<<<256, 512, 0, stream>>>(hm, bsum, bsq, N, HID);
  k_bnapply<<<cdiv((long long)N * HID, T), T, 0, stream>>>(hm, bsum, bsq, gmma, beta, N, HID);

  // 8) GEMM2 + layer-2 attention dots
  k_gemm2<<<cdiv((long long)N, 8), T, 0, stream>>>(hm, W2, as2, ad2, h2, oas2, oad2, N, HID, L);

  // 9) edge softmax, layer 2 (H=1)
  k_fillf<<<cdiv(N, T), T, 0, stream>>>(m2, -INFINITY, N);
  hipMemsetAsync(den2, 0, (size_t)N * 4, stream);
  k_edgeA<<<cdiv(E2, T), T, 0, stream>>>(ei, E, N, 1, oas2, oad2, e2, m2);
  k_edgeB<<<cdiv(E2, T), T, 0, stream>>>(ei, E, N, 1, m2, e2, den2);

  // 10) final aggregation + bias -> output
  k_agg2<<<cdiv((long long)N, 8), T, 0, stream>>>(h2, e2, den2, offv, csr, ei, b2, z, N, E, L);
}
